// RWKV_Time_Mixing_Block_73409581023491
// MI455X (gfx1250) — matrix operand order repaired and checked by reading, not compiled, not run on hardware
//
#include <hip/hip_runtime.h>

// ---------------------------------------------------------------------------
// RWKV time-mixing block for MI455X (gfx1250, wave32, WMMA bf16)
//   k = (x*mk + lx*(1-mk)) @ Wk ; v = ... @ Wv ; r = sigmoid(... @ Wr)
//   wkv/num/den epilogue fused into GEMM 1; hidden = rwkv @ Wout in GEMM 2.
// ---------------------------------------------------------------------------

typedef __bf16 bf16_t;
typedef __attribute__((ext_vector_type(16))) __bf16 v16bf;
typedef __attribute__((ext_vector_type(8)))  float  v8f;
typedef __attribute__((ext_vector_type(4)))  int    v4i;

#define DIM   1024
#define ATT   1024
#define BATCH 16384
#define KC    32      // WMMA K per step (bf16)
#define LDA   40      // shorts per LDS row: 32 data + 8 pad (80B: conflict-free b128)

#if defined(__gfx1250__) && __has_builtin(__builtin_amdgcn_global_load_async_to_lds_b128)
#define HAVE_ASYNC_LDS 1
typedef __attribute__((address_space(1))) v4i* gv4i_p;   // global int4*
typedef __attribute__((address_space(3))) v4i* lv4i_p;   // LDS int4*
#endif

__device__ __forceinline__ unsigned f2bf_bits(float f) {
    union { float f; unsigned u; } v; v.f = f;
    return (v.u + 0x7FFFu + ((v.u >> 16) & 1u)) >> 16;   // round-to-nearest-even
}
__device__ __forceinline__ unsigned pack_bf16(float a, float b) {
    return f2bf_bits(a) | (f2bf_bits(b) << 16);
}
__device__ __forceinline__ unsigned short f2bf(float f) {
    return (unsigned short)f2bf_bits(f);
}

union Frag { uint4 q[2]; v16bf v; };

// ---------------------------------------------------------------------------
// Kernel 1: fused mix + triple GEMM (k,v,r) + WKV epilogue.
// Block = 256 threads (8 waves). Tile 64(M) x 64(N). Wave (wm,wn): wm=wave&3
// owns 16 M-rows; wn=wave>>2 owns 32 N-cols (2 WMMA tiles) -> 6 accumulators.
// ---------------------------------------------------------------------------
__global__ __launch_bounds__(256)
void kvr_kernel(const float* __restrict__ x,    const float* __restrict__ lx,
                const float* __restrict__ lnum, const float* __restrict__ lden,
                const float* __restrict__ mixk, const float* __restrict__ mixv,
                const float* __restrict__ mixr, const float* __restrict__ dec,
                const float* __restrict__ bon,
                const float* __restrict__ Wk,   const float* __restrict__ Wv,
                const float* __restrict__ Wr,
                unsigned short* __restrict__ rwkv_ws,
                float* __restrict__ num_out,    float* __restrict__ den_out)
{
    __shared__ unsigned short lA[3][64 * LDA];   // mixed A tiles (bf16), [m][k]
    __shared__ unsigned short lB[3][64 * LDA];   // W tiles transposed (bf16), [n][k]

    const int t     = threadIdx.x;
    const int lane  = t & 31;
    const int wave  = t >> 5;
    const int wm    = wave & 3;
    const int wn    = wave >> 2;
    const int hi    = lane >> 4;
    const int lrow  = lane & 15;
    const int mBase = blockIdx.y * 64;
    const int nBase = blockIdx.x * 64;

    v8f acc[3][2] = {};   // [matrix k/v/r][n-tile]

    for (int k0 = 0; k0 < DIM; k0 += KC) {
        __syncthreads();

        // ---- fill A tiles: load x/last_x 64x32 f32, mix, convert to bf16 ----
        #pragma unroll
        for (int rep = 0; rep < 2; ++rep) {
            int f   = t + rep * 256;       // float4 slot 0..511
            int row = f >> 3;              // 0..63
            int kk  = (f & 7) * 4;         // 0..28
            size_t gi = (size_t)(mBase + row) * DIM + k0 + kk;
            float4 xv4 = *(const float4*)(x  + gi);
            float4 lv4 = *(const float4*)(lx + gi);
            if (rep == 0 && k0 + KC < DIM) {      // warm GL2/L0 for next K-chunk
                __builtin_prefetch(x  + gi + KC, 0, 1);
                __builtin_prefetch(lx + gi + KC, 0, 1);
            }
            float4 mk4 = *(const float4*)(mixk + k0 + kk);
            float4 mv4 = *(const float4*)(mixv + k0 + kk);
            float4 mr4 = *(const float4*)(mixr + k0 + kk);
            unsigned short* dk = &lA[0][row * LDA + kk];
            unsigned short* dv = &lA[1][row * LDA + kk];
            unsigned short* dr = &lA[2][row * LDA + kk];
#define MIXSTORE(dst, mm)                                                     \
            {                                                                 \
                float a0 = xv4.x * mm.x + lv4.x * (1.0f - mm.x);              \
                float a1 = xv4.y * mm.y + lv4.y * (1.0f - mm.y);              \
                float a2 = xv4.z * mm.z + lv4.z * (1.0f - mm.z);              \
                float a3 = xv4.w * mm.w + lv4.w * (1.0f - mm.w);              \
                uint2 p; p.x = pack_bf16(a0, a1); p.y = pack_bf16(a2, a3);    \
                *(uint2*)(dst) = p;                                           \
            }
            MIXSTORE(dk, mk4)
            MIXSTORE(dv, mv4)
            MIXSTORE(dr, mr4)
#undef MIXSTORE
        }

        // ---- fill B tiles: W[k0:k0+32, nBase:nBase+64] f32 -> bf16 transposed
        #pragma unroll
        for (int rep = 0; rep < 2; ++rep) {
            int f  = t + rep * 256;        // float4 slot 0..511
            int kk = f >> 4;               // 0..31 (K row)
            int nn = (f & 15) * 4;         // 0..60 (N col)
            size_t gi = (size_t)(k0 + kk) * ATT + nBase + nn;
            float4 wk4 = *(const float4*)(Wk + gi);
            float4 wv4 = *(const float4*)(Wv + gi);
            float4 wr4 = *(const float4*)(Wr + gi);
#define TSTORE(arr, w4)                                                       \
            arr[(nn + 0) * LDA + kk] = f2bf(w4.x);                            \
            arr[(nn + 1) * LDA + kk] = f2bf(w4.y);                            \
            arr[(nn + 2) * LDA + kk] = f2bf(w4.z);                            \
            arr[(nn + 3) * LDA + kk] = f2bf(w4.w);
            TSTORE(lB[0], wk4)
            TSTORE(lB[1], wv4)
            TSTORE(lB[2], wr4)
#undef TSTORE
        }
        __syncthreads();

        // ---- fragments + WMMA ----
        // A (16-bit 16x32): lane<16 -> K {0..7,16..23}, lane>=16 -> {8..15,24..31}
        Frag fa[3];
        #pragma unroll
        for (int m = 0; m < 3; ++m) {
            const char* ar = (const char*)&lA[m][(wm * 16 + lrow) * LDA];
            fa[m].q[0] = *(const uint4*)(ar + hi * 16);
            fa[m].q[1] = *(const uint4*)(ar + 32 + hi * 16);
        }
        // B (32x16): lane<16 -> N=lane, K {0..7,16..23} ; lane>=16 -> N=lane-16, K {8..15,24..31}
        #pragma unroll
        for (int nt = 0; nt < 2; ++nt) {
            #pragma unroll
            for (int m = 0; m < 3; ++m) {
                const char* br =
                    (const char*)&lB[m][(wn * 32 + nt * 16 + lrow) * LDA];
                Frag fb;
                fb.q[0] = *(const uint4*)(br + hi * 16);
                fb.q[1] = *(const uint4*)(br + 32 + hi * 16);
                acc[m][nt] = __builtin_amdgcn_wmma_f32_16x16x32_bf16(
                    false, fa[m].v, false, fb.v, (short)0, acc[m][nt],
                    false, false);
            }
        }
    }

    // ---- WKV epilogue: C/D layout: VGPR r -> M = r + 8*hi, N = lrow ----
    const int rowB = mBase + wm * 16 + hi * 8;
    #pragma unroll
    for (int nt = 0; nt < 2; ++nt) {
        const int col = nBase + wn * 32 + nt * 16 + lrow;
        const float bonus = bon[col];
        const float wdec  = __expf(-__expf(dec[col]));
        #pragma unroll
        for (int r = 0; r < 8; ++r) {
            const size_t idx = (size_t)(rowB + r) * ATT + col;
            const float kk = acc[0][nt][r];
            const float vv = acc[1][nt][r];
            const float rr = 1.0f / (1.0f + __expf(-acc[2][nt][r]));
            const float ln = lnum[idx];
            const float ld = lden[idx];
            const float ebk = __expf(bonus + kk);
            const float wkv = (ln + ebk * vv) / (ld + ebk);
            rwkv_ws[idx] = f2bf(rr * wkv);
            const float ek = __expf(kk);
            num_out[idx] = wdec * ln + ek * vv;
            den_out[idx] = wdec * ld + ek;
        }
    }
}

// ---------------------------------------------------------------------------
// Kernel 2: hidden = rwkv(bf16) @ Wout(f32 -> bf16).  Same tiling as kernel 1.
// A tile is a pure bf16 copy -> use async global->LDS (ASYNCcnt) when available.
// ---------------------------------------------------------------------------
__global__ __launch_bounds__(256)
void out_gemm(const unsigned short* __restrict__ rwkv,
              const float* __restrict__ Wout,
              float* __restrict__ hidden)
{
    __shared__ unsigned short lA[64 * LDA];
    __shared__ unsigned short lB[64 * LDA];

    const int t     = threadIdx.x;
    const int lane  = t & 31;
    const int wave  = t >> 5;
    const int wm    = wave & 3;
    const int wn    = wave >> 2;
    const int hi    = lane >> 4;
    const int lrow  = lane & 15;
    const int mBase = blockIdx.y * 64;
    const int nBase = blockIdx.x * 64;

    v8f acc[2] = {};

    for (int k0 = 0; k0 < ATT; k0 += KC) {
        __syncthreads();
        // A tile: already bf16 in workspace; each lane moves 16B (b128/lane).
        {
            int row = t >> 2;
            int kk  = (t & 3) * 8;
            const unsigned short* gp =
                rwkv + (size_t)(mBase + row) * ATT + k0 + kk;
            unsigned short* lp = &lA[row * LDA + kk];
#ifdef HAVE_ASYNC_LDS
            __builtin_amdgcn_global_load_async_to_lds_b128(
                (gv4i_p)(void*)gp, (lv4i_p)(void*)lp, 0, 0);
#else
            *(uint4*)(void*)lp = *(const uint4*)gp;
#endif
        }
        // B tile: Wout[k0:k0+32, nBase:nBase+64] f32 -> bf16 transposed.
        #pragma unroll
        for (int rep = 0; rep < 2; ++rep) {
            int f  = t + rep * 256;
            int kk = f >> 4;
            int nn = (f & 15) * 4;
            size_t gi = (size_t)(k0 + kk) * DIM + nBase + nn;
            float4 w4 = *(const float4*)(Wout + gi);
            if (rep == 0 && k0 + KC < ATT)
                __builtin_prefetch(Wout + gi + (size_t)KC * DIM, 0, 1);
            lB[(nn + 0) * LDA + kk] = f2bf(w4.x);
            lB[(nn + 1) * LDA + kk] = f2bf(w4.y);
            lB[(nn + 2) * LDA + kk] = f2bf(w4.z);
            lB[(nn + 3) * LDA + kk] = f2bf(w4.w);
        }
#ifdef HAVE_ASYNC_LDS
        asm volatile("s_wait_asynccnt 0x0" ::: "memory");  // A tile landed in LDS
#endif
        __syncthreads();

        Frag fa;
        const char* ar = (const char*)&lA[(wm * 16 + lrow) * LDA];
        fa.q[0] = *(const uint4*)(ar + hi * 16);
        fa.q[1] = *(const uint4*)(ar + 32 + hi * 16);
        #pragma unroll
        for (int nt = 0; nt < 2; ++nt) {
            const char* br = (const char*)&lB[(wn * 32 + nt * 16 + lrow) * LDA];
            Frag fb;
            fb.q[0] = *(const uint4*)(br + hi * 16);
            fb.q[1] = *(const uint4*)(br + 32 + hi * 16);
            acc[nt] = __builtin_amdgcn_wmma_f32_16x16x32_bf16(
                false, fa.v, false, fb.v, (short)0, acc[nt], false, false);
        }
    }

    const int rowB = mBase + wm * 16 + hi * 8;
    #pragma unroll
    for (int nt = 0; nt < 2; ++nt) {
        const int col = nBase + wn * 32 + nt * 16 + lrow;
        #pragma unroll
        for (int r = 0; r < 8; ++r)
            hidden[(size_t)(rowB + r) * DIM + col] = acc[nt][r];
    }
}

// ---------------------------------------------------------------------------
extern "C" void kernel_launch(void* const* d_in, const int* in_sizes, int n_in,
                              void* d_out, int out_size, void* d_ws,
                              size_t ws_size, hipStream_t stream)
{
    (void)in_sizes; (void)n_in; (void)out_size; (void)ws_size;

    const float* x     = (const float*)d_in[0];
    const float* lx    = (const float*)d_in[1];
    const float* lnum  = (const float*)d_in[2];
    const float* lden  = (const float*)d_in[3];
    const float* mixk  = (const float*)d_in[4];
    const float* mixv  = (const float*)d_in[5];
    const float* mixr  = (const float*)d_in[6];
    const float* dec   = (const float*)d_in[7];
    const float* bon   = (const float*)d_in[8];
    const float* Wk    = (const float*)d_in[9];
    const float* Wv    = (const float*)d_in[10];
    const float* Wr    = (const float*)d_in[11];
    const float* Wout  = (const float*)d_in[12];

    const size_t NEL = (size_t)BATCH * DIM;            // 16M elements per output
    float* hidden  = (float*)d_out;                    // out[0]
    float* out_x   = (float*)d_out + NEL;              // out[1] = x passthrough
    float* out_num = (float*)d_out + 2 * NEL;          // out[2]
    float* out_den = (float*)d_out + 3 * NEL;          // out[3]
    unsigned short* ws = (unsigned short*)d_ws;        // rwkv bf16 [B, ATT] = 32MB

    // x passthrough (graph-capture-safe async D2D copy)
    (void)hipMemcpyAsync(out_x, (const void*)x, NEL * sizeof(float),
                         hipMemcpyDeviceToDevice, stream);

    dim3 grid1(ATT / 64, BATCH / 64);   // (16, 256)
    kvr_kernel<<<grid1, 256, 0, stream>>>(x, lx, lnum, lden, mixk, mixv, mixr,
                                          dec, bon, Wk, Wv, Wr, ws,
                                          out_num, out_den);

    dim3 grid2(DIM / 64, BATCH / 64);   // (16, 256)
    out_gemm<<<grid2, 256, 0, stream>>>(ws, Wout, hidden);
}
